// LinearAttentionCell_38036230373788
// MI455X (gfx1250) — matrix operand order repaired and checked by reading, not compiled, not run on hardware
//
#include <hip/hip_runtime.h>
#include <hip/hip_bf16.h>

// ---------------------------------------------------------------------------
// LinearAttentionCell for MI455X (gfx1250, wave32, WMMA).
// B=4, T=2048, C=2048.  ~480 GFLOP vs ~200MB unique traffic -> compute bound.
// All 7 GEMMs on v_wmma_f32_16x16x32_bf16 (f32 accumulate).
// Round 3: every GEMM tile copy is a gfx1250 async global->LDS b128 copy
// (ASYNCcnt) -- transposes are done once at the producer (prep / epilogue
// LDS-bounce), never per-tile in the K loops.
// ---------------------------------------------------------------------------

typedef __attribute__((ext_vector_type(16))) __bf16 v16bf;
typedef __attribute__((ext_vector_type(8)))  float  v8f;

#define B_  4
#define T_  2048
#define C_  2048

static constexpr float DKs  = 0.02209708691207961f;  // sqrt(1/2048)
static constexpr float EPSs = 1e-8f;

// Block tile: 64 (M) x 128 (N) x 32 (K).  8 waves as 2x4, wave tile 32x32.
#define BM  64
#define BN  128
#define BK  32
#define LDA 40   // padded LDS stride in bf16 elems (80B rows, 16B-aligned)
#define TRS 72   // transpose-bounce stride (144B rows, 16B-aligned)

#define WMMA_BF16(a, b, c) \
    __builtin_amdgcn_wmma_f32_16x16x32_bf16(false, (a), false, (b), (short)0, (c), false, false)

// A fragment (16x32 bf16, lane r = row M): two contiguous 8-half runs
// K = hl*8..+7 and 16+hl*8..+7 (ISA 7.12.2) -> two ds_load_b128.
__device__ __forceinline__ v16bf frag_a(const __bf16* row, int hl) {
    union { v16bf v; uint4 q[2]; } f;
    f.q[0] = *(const uint4*)(row + hl * 8);
    f.q[1] = *(const uint4*)(row + 16 + hl * 8);
    return f.v;
}

// B fragment (32x16 bf16, lane r = col N): two contiguous 8-half runs
// K = hl*8..+7 and 16+hl*8..+7 -> two ds_load_b128.
__device__ __forceinline__ v16bf frag_b(const __bf16* row, int hl) {
    union { v16bf v; uint4 q[2]; } f;
    f.q[0] = *(const uint4*)(row + hl * 8);
    f.q[1] = *(const uint4*)(row + 16 + hl * 8);
    return f.v;
}

// gfx1250 async copy: 16B global -> LDS, tracked by ASYNCcnt, no VGPR staging.
// Generic LDS pointer truncated to 32 bits == LDS_BASE-relative byte offset.
__device__ __forceinline__ void async_copy_b128(const __bf16* g, __bf16* l) {
    uint64_t ga = (uint64_t)(uintptr_t)g;
    uint32_t la = (uint32_t)(uintptr_t)l;
    asm volatile("global_load_async_to_lds_b128 %0, %1, off"
                 :: "v"(la), "v"(ga) : "memory");
}
__device__ __forceinline__ void wait_async0() {
    asm volatile("s_wait_asynccnt 0x0" ::: "memory");
}

union PK8 { __bf16 h[8]; uint4 q; };

// ---------------------------------------------------------------------------
// Prep A: f32 -> bf16, vectorized (x only).
// ---------------------------------------------------------------------------
__global__ __launch_bounds__(256)
void cvt_bf16_kernel(const float* __restrict__ src, __bf16* __restrict__ dst,
                     int n4) {
    int i = blockIdx.x * blockDim.x + threadIdx.x;
    if (i < n4) {
        float4 f = ((const float4*)src)[i];
        union { __bf16 h[4]; uint2 u; } o;
        o.h[0] = (__bf16)f.x; o.h[1] = (__bf16)f.y;
        o.h[2] = (__bf16)f.z; o.h[3] = (__bf16)f.w;
        ((uint2*)dst)[i] = o.u;
    }
}

// ---------------------------------------------------------------------------
// Prep B: f32 -> bf16 + transpose (weights): dst[n][k] = src[k][n].
// 64x64 tile bounced through LDS; coalesced b128 on both global sides.
// ---------------------------------------------------------------------------
__global__ __launch_bounds__(256)
void cvtT_kernel(const float* __restrict__ src, __bf16* __restrict__ dst) {
    __shared__ __bf16 tile[64 * TRS];      // [col][row]
    const int tid = threadIdx.x;
    const int r0 = blockIdx.y * 64;        // src row (k)
    const int c0 = blockIdx.x * 64;        // src col (n)
#pragma unroll
    for (int j = 0; j < 4; ++j) {
        int ch = tid + j * 256;            // 1024 float4 chunks
        int rr = ch >> 4, c4 = (ch & 15) * 4;
        float4 f = *(const float4*)&src[(size_t)(r0 + rr) * C_ + c0 + c4];
        tile[(c4 + 0) * TRS + rr] = (__bf16)f.x;
        tile[(c4 + 1) * TRS + rr] = (__bf16)f.y;
        tile[(c4 + 2) * TRS + rr] = (__bf16)f.z;
        tile[(c4 + 3) * TRS + rr] = (__bf16)f.w;
    }
    __syncthreads();
#pragma unroll
    for (int j = 0; j < 2; ++j) {
        int ch = tid + j * 256;            // 512 uint4 chunks
        int cl = ch >> 3, r8 = (ch & 7) * 8;
        *(uint4*)&dst[(size_t)(c0 + cl) * C_ + r0 + r8] =
            *(const uint4*)&tile[cl * TRS + r8];
    }
}

// ---------------------------------------------------------------------------
// Kernel 1: fused q/k/v projection.  A = x [B*T, C] bf16, B = WqT/WkT/WvT
// [N, K] bf16.  All tiles layout-preserving -> all copies async b128.
// Epilogue: q,k -> elu(z*dk)+1; writes q,k,v row-major AND kT,vT (via LDS
// transpose bounce, coalesced b128 stores) for the downstream GEMMs.
// ---------------------------------------------------------------------------
__global__ __launch_bounds__(256)
void qkv_kernel(const __bf16* __restrict__ xh,
                const __bf16* __restrict__ WqT,
                const __bf16* __restrict__ WkT,
                const __bf16* __restrict__ WvT,
                __bf16* __restrict__ qf,
                __bf16* __restrict__ kf,
                __bf16* __restrict__ vf,
                __bf16* __restrict__ kT,
                __bf16* __restrict__ vT) {
    __shared__ __bf16 Alds[2][BM * LDA];
    __shared__ __bf16 Blds[2][3][BN * LDA];
    __shared__ __bf16 Tr[BN * TRS];

    const int tid  = threadIdx.x;
    const int lane = tid & 31;
    const int wid  = tid >> 5;
    const int hl   = lane >> 4;
    const int r    = lane & 15;
    const int wm   = (wid >> 2) * 32;
    const int wn   = (wid & 3)  * 32;
    const int m_base = blockIdx.y * BM;
    const int n_base = blockIdx.x * BN;

    const __bf16* Ws[3] = {WqT, WkT, WvT};

    // Copy shapes (all row-major): A 1 chunk, each B 2 chunks per thread.
    const int a_mm = tid >> 2, a_c8 = (tid & 3) * 8;
    v8f acc[3][2][2] = {};
    const int NT = C_ / BK;

    // Prologue: tile 0 into buffer 0 (async).
    async_copy_b128(&xh[(size_t)(m_base + a_mm) * C_ + a_c8],
                    &Alds[0][a_mm * LDA + a_c8]);
#pragma unroll
    for (int w = 0; w < 3; ++w)
#pragma unroll
        for (int j = 0; j < 2; ++j) {
            int ch = tid + j * 256;
            int nn = ch >> 2, c8 = (ch & 3) * 8;
            async_copy_b128(&Ws[w][(size_t)(n_base + nn) * C_ + c8],
                            &Blds[0][w][nn * LDA + c8]);
        }
    wait_async0();
    __syncthreads();

    for (int t = 0; t < NT; ++t) {
        const int cur = t & 1, nxt = cur ^ 1;
        const int k1 = (t + 1) * BK;
        if (t + 1 < NT) {
            async_copy_b128(&xh[(size_t)(m_base + a_mm) * C_ + k1 + a_c8],
                            &Alds[nxt][a_mm * LDA + a_c8]);
#pragma unroll
            for (int w = 0; w < 3; ++w)
#pragma unroll
                for (int j = 0; j < 2; ++j) {
                    int ch = tid + j * 256;
                    int nn = ch >> 2, c8 = (ch & 3) * 8;
                    async_copy_b128(&Ws[w][(size_t)(n_base + nn) * C_ + k1 + c8],
                                    &Blds[nxt][w][nn * LDA + c8]);
                }
        }

        v16bf af[2], bfr[3][2];
#pragma unroll
        for (int mt = 0; mt < 2; ++mt)
            af[mt] = frag_a(&Alds[cur][(wm + mt * 16 + r) * LDA], hl);
#pragma unroll
        for (int w = 0; w < 3; ++w)
#pragma unroll
            for (int nt = 0; nt < 2; ++nt)
                bfr[w][nt] = frag_b(&Blds[cur][w][(wn + nt * 16 + r) * LDA], hl);
#pragma unroll
        for (int w = 0; w < 3; ++w)
#pragma unroll
            for (int mt = 0; mt < 2; ++mt)
#pragma unroll
                for (int nt = 0; nt < 2; ++nt)
                    acc[w][mt][nt] = WMMA_BF16(af[mt], bfr[w][nt], acc[w][mt][nt]);

        if (t + 1 < NT) wait_async0();
        __syncthreads();
    }

    // ---- Epilogue phase 1: row-major q, k, v (activated). ----
#pragma unroll
    for (int mt = 0; mt < 2; ++mt)
#pragma unroll
        for (int nt = 0; nt < 2; ++nt)
#pragma unroll
            for (int j = 0; j < 8; ++j) {
                int m = m_base + wm + mt * 16 + hl * 8 + j;
                int n = n_base + wn + nt * 16 + r;
                size_t o = (size_t)m * C_ + n;
                float zq = acc[0][mt][nt][j] * DKs;
                float zk = acc[1][mt][nt][j] * DKs;
                qf[o] = (__bf16)(zq > 0.f ? zq + 1.f : __expf(zq));  // elu+1
                kf[o] = (__bf16)(zk > 0.f ? zk + 1.f : __expf(zk));
                vf[o] = (__bf16)acc[2][mt][nt][j];
            }

    // ---- Epilogue phase 2: kT via LDS bounce (lane holds 8 consec M). ----
    __syncthreads();
#pragma unroll
    for (int mt = 0; mt < 2; ++mt)
#pragma unroll
        for (int nt = 0; nt < 2; ++nt) {
            PK8 p;
#pragma unroll
            for (int j = 0; j < 8; ++j) {
                float zk = acc[1][mt][nt][j] * DKs;
                p.h[j] = (__bf16)(zk > 0.f ? zk + 1.f : __expf(zk));
            }
            *(uint4*)&Tr[(wn + nt * 16 + r) * TRS + wm + mt * 16 + hl * 8] = p.q;
        }
    __syncthreads();
#pragma unroll
    for (int j = 0; j < 4; ++j) {
        int ch = tid + j * 256;                  // 1024 uint4 chunks
        int nl = ch >> 3, m8 = (ch & 7) * 8;
        *(uint4*)&kT[(size_t)(n_base + nl) * T_ + m_base + m8] =
            *(const uint4*)&Tr[nl * TRS + m8];
    }

    // ---- Epilogue phase 3: vT via the same bounce buffer. ----
    __syncthreads();
#pragma unroll
    for (int mt = 0; mt < 2; ++mt)
#pragma unroll
        for (int nt = 0; nt < 2; ++nt) {
            PK8 p;
#pragma unroll
            for (int j = 0; j < 8; ++j) p.h[j] = (__bf16)acc[2][mt][nt][j];
            *(uint4*)&Tr[(wn + nt * 16 + r) * TRS + wm + mt * 16 + hl * 8] = p.q;
        }
    __syncthreads();
#pragma unroll
    for (int j = 0; j < 4; ++j) {
        int ch = tid + j * 256;
        int nl = ch >> 3, m8 = (ch & 7) * 8;
        *(uint4*)&vT[(size_t)(n_base + nl) * T_ + m_base + m8] =
            *(const uint4*)&Tr[nl * TRS + m8];
    }
}

// ---------------------------------------------------------------------------
// Kernel 2: memory[b] = k[b]^T @ v[b]  (M=N=C, K=T).  Reads kT [C,T] and
// vT [C,T] -> both tiles row-major -> all async copies.  Writes memT
// (memory transposed) via the LDS bounce so kernel 3 gets a row-major B.
// ---------------------------------------------------------------------------
__global__ __launch_bounds__(256)
void memory_kernel(const __bf16* __restrict__ kT,
                   const __bf16* __restrict__ vT,
                   __bf16* __restrict__ memT) {
    __shared__ __bf16 Alds[2][BM * LDA];
    __shared__ __bf16 Blds[2][BN * LDA];
    __shared__ __bf16 Tr[BN * TRS];

    const int tid  = threadIdx.x;
    const int lane = tid & 31;
    const int wid  = tid >> 5;
    const int hl   = lane >> 4;
    const int r    = lane & 15;
    const int wm   = (wid >> 2) * 32;
    const int wn   = (wid & 3)  * 32;
    const int b      = blockIdx.z;
    const int m_base = blockIdx.y * BM;
    const int n_base = blockIdx.x * BN;

    const __bf16* kTb = kT + (size_t)b * C_ * T_;
    const __bf16* vTb = vT + (size_t)b * C_ * T_;

    const int a_mm = tid >> 2, a_c8 = (tid & 3) * 8;
    v8f acc[2][2] = {};
    const int NT = T_ / BK;

    async_copy_b128(&kTb[(size_t)(m_base + a_mm) * T_ + a_c8],
                    &Alds[0][a_mm * LDA + a_c8]);
#pragma unroll
    for (int j = 0; j < 2; ++j) {
        int ch = tid + j * 256;
        int nn = ch >> 2, c8 = (ch & 3) * 8;
        async_copy_b128(&vTb[(size_t)(n_base + nn) * T_ + c8],
                        &Blds[0][nn * LDA + c8]);
    }
    wait_async0();
    __syncthreads();

    for (int t = 0; t < NT; ++t) {
        const int cur = t & 1, nxt = cur ^ 1;
        const int k1 = (t + 1) * BK;
        if (t + 1 < NT) {
            async_copy_b128(&kTb[(size_t)(m_base + a_mm) * T_ + k1 + a_c8],
                            &Alds[nxt][a_mm * LDA + a_c8]);
#pragma unroll
            for (int j = 0; j < 2; ++j) {
                int ch = tid + j * 256;
                int nn = ch >> 2, c8 = (ch & 3) * 8;
                async_copy_b128(&vTb[(size_t)(n_base + nn) * T_ + k1 + c8],
                                &Blds[nxt][nn * LDA + c8]);
            }
        }

        v16bf af[2], bfr[2];
#pragma unroll
        for (int mt = 0; mt < 2; ++mt)
            af[mt] = frag_a(&Alds[cur][(wm + mt * 16 + r) * LDA], hl);
#pragma unroll
        for (int nt = 0; nt < 2; ++nt)
            bfr[nt] = frag_b(&Blds[cur][(wn + nt * 16 + r) * LDA], hl);
#pragma unroll
        for (int mt = 0; mt < 2; ++mt)
#pragma unroll
            for (int nt = 0; nt < 2; ++nt)
                acc[mt][nt] = WMMA_BF16(af[mt], bfr[nt], acc[mt][nt]);

        if (t + 1 < NT) wait_async0();
        __syncthreads();
    }

    // Epilogue: write memT[n][m] via bounce (coalesced b128 stores).
    __syncthreads();
#pragma unroll
    for (int mt = 0; mt < 2; ++mt)
#pragma unroll
        for (int nt = 0; nt < 2; ++nt) {
            PK8 p;
#pragma unroll
            for (int j = 0; j < 8; ++j) p.h[j] = (__bf16)acc[mt][nt][j];
            *(uint4*)&Tr[(wn + nt * 16 + r) * TRS + wm + mt * 16 + hl * 8] = p.q;
        }
    __syncthreads();
#pragma unroll
    for (int j = 0; j < 4; ++j) {
        int ch = tid + j * 256;
        int nl = ch >> 3, m8 = (ch & 7) * 8;
        *(uint4*)&memT[((size_t)b * C_ + n_base + nl) * C_ + m_base + m8] =
            *(const uint4*)&Tr[nl * TRS + m8];
    }
}

// ---------------------------------------------------------------------------
// Kernel 3: fused output.  Three GEMMs share one K loop (K = C):
//   accS = q @ memory (B from memT)   accW = v @ Wo (B from WoT)
//   accL = q @ k^T    (B from k row-major)
// All five tiles row-major -> all copies async b128.
// Epilogue: out = (accS + tanh(accW + bo)) / (accL + eps), f32.
// ---------------------------------------------------------------------------
__global__ __launch_bounds__(256)
void out_kernel(const __bf16* __restrict__ qf,
                const __bf16* __restrict__ kf,
                const __bf16* __restrict__ vf,
                const __bf16* __restrict__ memT,
                const __bf16* __restrict__ WoT,
                const float* __restrict__ bo,
                float* __restrict__ out) {
    __shared__ __bf16 Aq[2][BM * LDA];
    __shared__ __bf16 Av[2][BM * LDA];
    __shared__ __bf16 Bm[2][BN * LDA];
    __shared__ __bf16 Bw[2][BN * LDA];
    __shared__ __bf16 Bk[2][BN * LDA];

    const int tid  = threadIdx.x;
    const int lane = tid & 31;
    const int wid  = tid >> 5;
    const int hl   = lane >> 4;
    const int r    = lane & 15;
    const int wm   = (wid >> 2) * 32;
    const int wn   = (wid & 3)  * 32;
    const int b      = blockIdx.z;
    const int m_base = blockIdx.y * BM;
    const int n_base = blockIdx.x * BN;

    const __bf16* qb = qf + (size_t)b * T_ * C_;
    const __bf16* kb = kf + (size_t)b * T_ * C_;
    const __bf16* vb = vf + (size_t)b * T_ * C_;
    const __bf16* mb = memT + (size_t)b * C_ * C_;

    const int a_mm = tid >> 2, a_c8 = (tid & 3) * 8;
    v8f accS[2][2] = {}, accW[2][2] = {}, accL[2][2] = {};
    const int NT = C_ / BK;

    // Prologue (tile 0, buffer 0): 8 async copies per thread.
    async_copy_b128(&qb[(size_t)(m_base + a_mm) * C_ + a_c8],
                    &Aq[0][a_mm * LDA + a_c8]);
    async_copy_b128(&vb[(size_t)(m_base + a_mm) * C_ + a_c8],
                    &Av[0][a_mm * LDA + a_c8]);
#pragma unroll
    for (int j = 0; j < 2; ++j) {
        int ch = tid + j * 256;
        int nn = ch >> 2, c8 = (ch & 3) * 8;
        async_copy_b128(&kb[(size_t)(n_base + nn) * C_ + c8],
                        &Bk[0][nn * LDA + c8]);
        async_copy_b128(&mb[(size_t)(n_base + nn) * C_ + c8],
                        &Bm[0][nn * LDA + c8]);
        async_copy_b128(&WoT[(size_t)(n_base + nn) * C_ + c8],
                        &Bw[0][nn * LDA + c8]);
    }
    wait_async0();
    __syncthreads();

    for (int t = 0; t < NT; ++t) {
        const int cur = t & 1, nxt = cur ^ 1;
        const int k1 = (t + 1) * BK;
        if (t + 1 < NT) {
            async_copy_b128(&qb[(size_t)(m_base + a_mm) * C_ + k1 + a_c8],
                            &Aq[nxt][a_mm * LDA + a_c8]);
            async_copy_b128(&vb[(size_t)(m_base + a_mm) * C_ + k1 + a_c8],
                            &Av[nxt][a_mm * LDA + a_c8]);
#pragma unroll
            for (int j = 0; j < 2; ++j) {
                int ch = tid + j * 256;
                int nn = ch >> 2, c8 = (ch & 3) * 8;
                async_copy_b128(&kb[(size_t)(n_base + nn) * C_ + k1 + c8],
                                &Bk[nxt][nn * LDA + c8]);
                async_copy_b128(&mb[(size_t)(n_base + nn) * C_ + k1 + c8],
                                &Bm[nxt][nn * LDA + c8]);
                async_copy_b128(&WoT[(size_t)(n_base + nn) * C_ + k1 + c8],
                                &Bw[nxt][nn * LDA + c8]);
            }
        }

        v16bf aq[2], av[2], bm[2], bw[2], bk[2];
#pragma unroll
        for (int mt = 0; mt < 2; ++mt) {
            aq[mt] = frag_a(&Aq[cur][(wm + mt * 16 + r) * LDA], hl);
            av[mt] = frag_a(&Av[cur][(wm + mt * 16 + r) * LDA], hl);
        }
#pragma unroll
        for (int nt = 0; nt < 2; ++nt) {
            bm[nt] = frag_b(&Bm[cur][(wn + nt * 16 + r) * LDA], hl);
            bw[nt] = frag_b(&Bw[cur][(wn + nt * 16 + r) * LDA], hl);
            bk[nt] = frag_b(&Bk[cur][(wn + nt * 16 + r) * LDA], hl);
        }
#pragma unroll
        for (int mt = 0; mt < 2; ++mt)
#pragma unroll
            for (int nt = 0; nt < 2; ++nt) {
                accS[mt][nt] = WMMA_BF16(aq[mt], bm[nt], accS[mt][nt]);
                accW[mt][nt] = WMMA_BF16(av[mt], bw[nt], accW[mt][nt]);
                accL[mt][nt] = WMMA_BF16(aq[mt], bk[nt], accL[mt][nt]);
            }

        if (t + 1 < NT) wait_async0();
        __syncthreads();
    }

#pragma unroll
    for (int mt = 0; mt < 2; ++mt)
#pragma unroll
        for (int nt = 0; nt < 2; ++nt)
#pragma unroll
            for (int j = 0; j < 8; ++j) {
                int m = m_base + wm + mt * 16 + hl * 8 + j;
                int n = n_base + wn + nt * 16 + r;
                float s = accS[mt][nt][j] + tanhf(accW[mt][nt][j] + bo[n]);
                out[((size_t)b * T_ + m) * C_ + n] = s / (accL[mt][nt][j] + EPSs);
            }
}

// ---------------------------------------------------------------------------
// Launcher.  Workspace (bf16): xh 32MB | WqT/WkT/WvT/WoT 8MB ea |
// q,k,v 32MB ea | kT,vT 32MB ea | memT 32MB  = 256MB total.
// ---------------------------------------------------------------------------
extern "C" void kernel_launch(void* const* d_in, const int* in_sizes, int n_in,
                              void* d_out, int out_size, void* d_ws, size_t ws_size,
                              hipStream_t stream) {
    const float* x  = (const float*)d_in[0];
    const float* Wq = (const float*)d_in[1];
    const float* Wk = (const float*)d_in[2];
    const float* Wv = (const float*)d_in[3];
    const float* Wo = (const float*)d_in[4];
    const float* bo = (const float*)d_in[5];
    float* out = (float*)d_out;

    const size_t nX = (size_t)B_ * T_ * C_;   // == B*C*C since T == C
    const size_t nW = (size_t)C_ * C_;

    __bf16* xh   = (__bf16*)d_ws;
    __bf16* WqT  = xh + nX;
    __bf16* WkT  = WqT + nW;
    __bf16* WvT  = WkT + nW;
    __bf16* WoT  = WvT + nW;
    __bf16* qfp  = WoT + nW;
    __bf16* kfp  = qfp + nX;
    __bf16* vfp  = kfp + nX;
    __bf16* kTp  = vfp + nX;
    __bf16* vTp  = kTp + nX;
    __bf16* memT = vTp + nX;

    dim3 blk(256);
    // Prep: bf16-convert x; bf16-convert + transpose the four weights.
    cvt_bf16_kernel<<<(int)(nX / 4 + 255) / 256, blk, 0, stream>>>(x, xh, (int)(nX / 4));
    dim3 tg(C_ / 64, C_ / 64);
    cvtT_kernel<<<tg, blk, 0, stream>>>(Wq, WqT);
    cvtT_kernel<<<tg, blk, 0, stream>>>(Wk, WkT);
    cvtT_kernel<<<tg, blk, 0, stream>>>(Wv, WvT);
    cvtT_kernel<<<tg, blk, 0, stream>>>(Wo, WoT);

    qkv_kernel<<<dim3(C_ / BN, (B_ * T_) / BM), blk, 0, stream>>>(
        xh, WqT, WkT, WvT, qfp, kfp, vfp, kTp, vTp);
    memory_kernel<<<dim3(C_ / BN, C_ / BM, B_), blk, 0, stream>>>(
        kTp, vTp, memT);
    out_kernel<<<dim3(C_ / BN, T_ / BM, B_), blk, 0, stream>>>(
        qfp, kfp, vfp, memT, WoT, bo, out);
}
